// GCNModel_84215718740073
// MI455X (gfx1250) — matrix operand order repaired and checked by reading, not compiled, not run on hardware
//
#include <hip/hip_runtime.h>

typedef __attribute__((ext_vector_type(16))) _Float16 v16h;
typedef __attribute__((ext_vector_type(8)))  _Float16 v8h;
typedef __attribute__((ext_vector_type(8)))  float    v8f;

// ---------------- GCN normalization ----------------

__global__ __launch_bounds__(256) void k_fill1(float* p, int n) {
  int i = blockIdx.x * blockDim.x + threadIdx.x;
  if (i < n) p[i] = 1.0f;  // self-loop weight
}

__global__ __launch_bounds__(256) void k_deg_edges(const int* __restrict__ row,
    const float* __restrict__ w, float* __restrict__ deg, int E) {
  int e = blockIdx.x * blockDim.x + threadIdx.x;
  if (e < E) atomicAdd(&deg[row[e]], w[e]);
}

__global__ __launch_bounds__(256) void k_dinv(float* deg, int n) {
  int i = blockIdx.x * blockDim.x + threadIdx.x;
  if (i < n) {
    float d = deg[i];
    deg[i] = (d > 0.0f) ? rsqrtf(fmaxf(d, 1e-12f)) : 0.0f;
  }
}

__global__ __launch_bounds__(256) void k_norm(const int* __restrict__ row,
    const int* __restrict__ col, const float* __restrict__ w,
    const float* __restrict__ dinv, float* __restrict__ nrm, int E) {
  int e = blockIdx.x * blockDim.x + threadIdx.x;
  if (e < E) nrm[e] = dinv[row[e]] * w[e] * dinv[col[e]];
}

// ---------------- precision / layout converters ----------------

__global__ __launch_bounds__(256) void k_f32_to_f16(const float* __restrict__ in,
    _Float16* __restrict__ out, int n) {
  int i = blockIdx.x * blockDim.x + threadIdx.x;
  if (i < n) out[i] = (_Float16)in[i];
}

// W [K x F] row-major f32  ->  Wt [Fpad x Kpad] row-major f16 (zero padded)
__global__ __launch_bounds__(256) void k_wt(const float* __restrict__ W,
    _Float16* __restrict__ Wt, int K, int F, int Kpad, int Fpad) {
  int i = blockIdx.x * blockDim.x + threadIdx.x;
  int tot = Kpad * Fpad;
  if (i < tot) {
    int o = i / Kpad, k = i - o * Kpad;
    Wt[i] = (o < F && k < K) ? (_Float16)W[k * F + o] : (_Float16)0.0f;
  }
}

// ---------------- WMMA GEMM: C[N x ldc] = A[N x K](f16) * Wt^T ----------------
// one wave per 16x16 output tile; contiguous-k register tiles (k = 16*half + s), EXEC all-1s.

__global__ __launch_bounds__(256) void k_gemm(const _Float16* __restrict__ A,
    const _Float16* __restrict__ Bt, float* __restrict__ C,
    int K, int ldc, int ntilesN, int ntiles) {
  int wid  = blockIdx.x * (blockDim.x >> 5) + (threadIdx.x >> 5);  // uniform per wave
  if (wid >= ntiles) return;
  int lane = threadIdx.x & 31;
  int rt = wid / ntilesN;
  int ct = wid - rt * ntilesN;
  int half = lane >> 4;      // 0: lanes 0-15, 1: lanes 16-31
  int l15  = lane & 15;

  // A: lane holds row rt*16+l15; K slots contiguous 16, base 16*half
  const _Float16* ap = A  + (size_t)(rt * 16 + l15) * K + half * 16;
  // B (=Wt row = column of W): lane holds col ct*16+l15; K slots contiguous 16, base 16*half
  const _Float16* bp = Bt + (size_t)(ct * 16 + l15) * K + half * 16;

  v8f acc = {};
  for (int k0 = 0; k0 < K; k0 += 32) {
    v8h alo = *(const v8h*)(ap + k0);
    v8h ahi = *(const v8h*)(ap + k0 + 8);
    v16h a = __builtin_shufflevector(alo, ahi, 0,1,2,3,4,5,6,7,8,9,10,11,12,13,14,15);
    v16h b = *(const v16h*)(bp + k0);
    acc = __builtin_amdgcn_wmma_f32_16x16x32_f16(
        /*neg_a=*/false, a, /*neg_b=*/false, b,
        /*c_mod=*/(short)0, acc, /*reuse_a=*/false, /*reuse_b=*/false);
  }
  // D layout: VGPR r -> M = r + 8*half, N = l15
  int mbase = rt * 16 + half * 8;
  float* cp = C + (size_t)mbase * ldc + ct * 16 + l15;
#pragma unroll
  for (int r = 0; r < 8; r++) cp[(size_t)r * ldc] = acc[r];
}

// ---------------- aggregation ----------------

// agg[n][f] = bias[f] + dinv[n]^2 * h[n][f]   (self loop term + bias)
__global__ __launch_bounds__(256) void k_self_init(const float* __restrict__ h,
    const float* __restrict__ dinv, const float* __restrict__ bias,
    float* __restrict__ agg, int N, int F, int ld) {
  int i = blockIdx.x * blockDim.x + threadIdx.x;
  int tot = N * F;
  if (i < tot) {
    int n = i / F, f = i - n * F;
    float s = dinv[n];
    agg[(size_t)n * ld + f] = bias[f] + s * s * h[(size_t)n * ld + f];
  }
}

// one wave per edge: agg[row] += norm * h[col]   (atomics resolve in L2; h fits in 192MB L2)
__global__ __launch_bounds__(256) void k_edge_agg(const int* __restrict__ row,
    const int* __restrict__ col, const float* __restrict__ nrm,
    const float* __restrict__ h, float* __restrict__ agg, int E, int F, int ld) {
  int wid  = blockIdx.x * (blockDim.x >> 5) + (threadIdx.x >> 5);
  int lane = threadIdx.x & 31;
  if (wid >= E) return;
  int r = row[wid], c = col[wid];
  float wv = nrm[wid];
  const float* hs = h   + (size_t)c * ld;
  float*       ad = agg + (size_t)r * ld;
  for (int f = lane; f < F; f += 32) atomicAdd(&ad[f], wv * hs[f]);
}

// relu + f16 repack (zero pad) as next layer's WMMA A-matrix
__global__ __launch_bounds__(256) void k_relu_f16(const float* __restrict__ agg,
    _Float16* __restrict__ out, int N, int F, int ld, int ldo) {
  int i = blockIdx.x * blockDim.x + threadIdx.x;
  int tot = N * ldo;
  if (i < tot) {
    int n = i / ldo, f = i - n * ldo;
    float v = (f < F) ? fmaxf(agg[(size_t)n * ld + f], 0.0f) : 0.0f;
    out[i] = (_Float16)v;
  }
}

// last layer: no activation, compact f32 write to d_out
__global__ __launch_bounds__(256) void k_final_out(const float* __restrict__ agg,
    float* __restrict__ out, int N, int F, int ld) {
  int i = blockIdx.x * blockDim.x + threadIdx.x;
  int tot = N * F;
  if (i < tot) {
    int n = i / F, f = i - n * F;
    out[i] = agg[(size_t)n * ld + f];
  }
}

// ---------------- driver ----------------

extern "C" void kernel_launch(void* const* d_in, const int* in_sizes, int n_in,
                              void* d_out, int out_size, void* d_ws, size_t ws_size,
                              hipStream_t stream) {
  (void)n_in; (void)out_size; (void)ws_size;
  const float* x  = (const float*)d_in[0];
  const int*   ei = (const int*)d_in[1];
  const float* ew = (const float*)d_in[2];
  const float* W[4] = {(const float*)d_in[3], (const float*)d_in[5],
                       (const float*)d_in[7], (const float*)d_in[9]};
  const float* B[4] = {(const float*)d_in[4], (const float*)d_in[6],
                       (const float*)d_in[8], (const float*)d_in[10]};
  const int N = in_sizes[0] / 128;   // 50000 (multiple of 16)
  const int E = in_sizes[2];         // 800000
  const int* row = ei;
  const int* col = ei + E;

  const int Fin [4] = {128, 100, 100, 50};
  const int Fout[4] = {100, 100,  50,  6};
  const int Kpad[4] = {128, 128, 128, 64};   // GEMM K (padded input feats)
  const int Fpad[4] = {112, 112,  64, 16};   // GEMM N (padded output feats)

  // carve workspace
  char* wsp = (char*)d_ws;
  auto carve = [&](size_t bytes) { char* p = wsp; wsp += (bytes + 255) & ~size_t(255); return p; };
  float*    deg  = (float*)   carve((size_t)N * 4);
  float*    nrm  = (float*)   carve((size_t)E * 4);
  _Float16* bufA = (_Float16*)carve((size_t)N * 128 * 2);
  _Float16* bufB = (_Float16*)carve((size_t)N * 128 * 2);
  float*    h    = (float*)   carve((size_t)N * 112 * 4);
  float*    agg  = (float*)   carve((size_t)N * 112 * 4);
  _Float16* Wt[4];
  for (int l = 0; l < 4; l++) Wt[l] = (_Float16*)carve((size_t)Fpad[l] * Kpad[l] * 2);

  auto cdiv = [](int a, int b) { return (a + b - 1) / b; };

  // norm coefficients (shared across layers, like the reference)
  k_fill1    <<<cdiv(N, 256), 256, 0, stream>>>(deg, N);
  k_deg_edges<<<cdiv(E, 256), 256, 0, stream>>>(row, ew, deg, E);
  k_dinv     <<<cdiv(N, 256), 256, 0, stream>>>(deg, N);   // deg -> dinv in place
  k_norm     <<<cdiv(E, 256), 256, 0, stream>>>(row, col, ew, deg, nrm, E);

  // inputs to f16 / transposed-padded weights
  k_f32_to_f16<<<cdiv(N * 128, 256), 256, 0, stream>>>(x, bufA, N * 128);
  for (int l = 0; l < 4; l++)
    k_wt<<<cdiv(Fpad[l] * Kpad[l], 256), 256, 0, stream>>>(W[l], Wt[l], Fin[l], Fout[l], Kpad[l], Fpad[l]);

  _Float16* in16  = bufA;
  _Float16* out16 = bufB;
  for (int l = 0; l < 4; l++) {
    int ntN    = Fpad[l] / 16;
    int ntiles = (N / 16) * ntN;
    k_gemm<<<cdiv(ntiles, 8), 256, 0, stream>>>(in16, Wt[l], h, Kpad[l], Fpad[l], ntN, ntiles);
    k_self_init<<<cdiv(N * Fout[l], 256), 256, 0, stream>>>(h, deg, B[l], agg, N, Fout[l], Fpad[l]);
    k_edge_agg <<<cdiv(E, 8), 256, 0, stream>>>(row, col, nrm, h, agg, E, Fout[l], Fpad[l]);
    if (l < 3) {
      int ldo = Kpad[l + 1];
      k_relu_f16<<<cdiv(N * ldo, 256), 256, 0, stream>>>(agg, out16, N, Fout[l], Fpad[l], ldo);
      _Float16* t = in16; in16 = out16; out16 = t;
    } else {
      k_final_out<<<cdiv(N * 6, 256), 256, 0, stream>>>(agg, (float*)d_out, N, 6, Fpad[l]);
    }
  }
}
